// GATv2_5171140625164
// MI455X (gfx1250) — matrix operand order repaired and checked by reading, not compiled, not run on hardware
//
#include <hip/hip_runtime.h>
#include <hip/hip_bf16.h>

// ---------------------------------------------------------------------------
// GATv2 3-layer network, MI455X (gfx1250, wave32, WMMA).
//  - Linear transforms + readout GEMM: v_wmma_f32_16x16x32_bf16, fp32 accum.
//  - Attention scores+softmax: fused VALU kernel (leaky-relu inside the dot is
//    elementwise, not a matmul), alpha emitted as bf16.
//  - Aggregation out = alpha @ xl: batched WMMA GEMM per (b,h); B tile staged
//    into LDS via global_load_async_to_lds_b128 (ASYNCcnt) for reuse by all
//    8 waves; bias+relu fused into the epilogue.
// ---------------------------------------------------------------------------

typedef __attribute__((ext_vector_type(16))) __bf16 v16bf;
typedef __attribute__((ext_vector_type(8)))  __bf16 v8bf;
typedef __attribute__((ext_vector_type(8)))  float  v8f;

#define MDIM 2048   // B*N
#define NNODES 512
#define NBATCH 4

// ---------------- pack helpers ----------------
__global__ void pack_bf16_kernel(const float* __restrict__ x, __bf16* __restrict__ y, int n) {
    int i = blockIdx.x * blockDim.x + threadIdx.x;
    if (i < n) y[i] = (__bf16)x[i];
}

// W [K x N] fp32  ->  WT [N x K] bf16
__global__ void pack_wt_kernel(const float* __restrict__ W, __bf16* __restrict__ WT, int K, int N) {
    int i = blockIdx.x * blockDim.x + threadIdx.x;
    if (i < K * N) {
        int n = i / K, k = i - n * K;
        WT[(size_t)n * K + k] = (__bf16)W[(size_t)k * N + n];
    }
}

// xl [B*N x HC] fp32 -> xlT [(b*H+h)*C + c][j] bf16   (j contiguous)
__global__ void pack_xlt_kernel(const float* __restrict__ xl, __bf16* __restrict__ xlT,
                                int H, int C) {
    int HC = H * C;
    int idx = blockIdx.x * blockDim.x + threadIdx.x;      // dest-linear
    if (idx < NBATCH * HC * NNODES) {
        int j = idx & (NNODES - 1);
        int r = idx >> 9;           // (b*H+h)*C + c
        int c = r % C;
        int h = (r / C) % H;
        int b = r / (C * H);
        xlT[idx] = (__bf16)xl[((size_t)b * NNODES + j) * HC + h * C + c];
    }
}

// ---------------- WMMA GEMM: C[MxN] = A[MxK](bf16) * W (via WT[NxK] bf16) ----
__global__ __launch_bounds__(128)
void gemm_bf16_kernel(const __bf16* __restrict__ A, const __bf16* __restrict__ WT,
                      const float* __restrict__ bias, float* __restrict__ C,
                      int M, int K, int N, int addBias)
{
    const int t    = threadIdx.x;
    const int lane = t & 31;
    const int wid  = blockIdx.x * (blockDim.x >> 5) + (t >> 5);
    const int ntiles = N >> 4;
    const int tm = (wid / ntiles) << 4;
    const int tn = (wid % ntiles) << 4;
    const int h   = lane >> 4;
    const int r16 = lane & 15;

    v8f acc = {};
    const __bf16* arow = A  + (size_t)(tm + r16) * K;
    const __bf16* brow = WT + (size_t)(tn + r16) * K;

    for (int kb = 0; kb < K; kb += 32) {
        v8bf a0 = *(const v8bf*)(arow + kb + 16 * h);
        v8bf a1 = *(const v8bf*)(arow + kb + 16 * h + 8);
        v16bf av;
#pragma unroll
        for (int i = 0; i < 8; ++i) { av[i] = a0[i]; av[8 + i] = a1[i]; }
        v16bf bv = *(const v16bf*)(brow + kb + 16 * h);
        acc = __builtin_amdgcn_wmma_f32_16x16x32_bf16(
                  false, av, false, bv, (short)0, acc, false, false);
    }

    float badd = addBias ? bias[tn + r16] : 0.0f;
    float* cbase = C + (size_t)tm * N + tn;
#pragma unroll
    for (int r = 0; r < 8; ++r)
        cbase[(size_t)(r + 8 * h) * N + r16] = acc[r] + badd;
}

// ---------------- scores + softmax -> alpha (bf16) --------------------------
// One block per (i, b). Block = H*C threads. e kept in LDS only.
template <int H, int C>
__global__ __launch_bounds__(H * C)
void score_kernel(const float* __restrict__ xl, const float* __restrict__ xr,
                  const int* __restrict__ adj, const float* __restrict__ att,
                  __bf16* __restrict__ alpha)
{
    constexpr int HC = H * C;
    constexpr int NW = HC / 32;
    constexpr int KP = HC / 32;
    const int t    = threadIdx.x;
    const int lane = t & 31;
    const int w    = t >> 5;
    const int i    = blockIdx.x;
    const int b    = blockIdx.y;

    __shared__ float e_sh[H][NNODES];
    __shared__ float wred[NW][H];
    __shared__ float smax[H];
    __shared__ float ssum[H];

    const size_t rowI = (size_t)b * NNODES + i;

    float xr_v[KP], att_v[KP];
#pragma unroll
    for (int k = 0; k < KP; ++k) {
        xr_v[k]  = xr[rowI * HC + 32 * k + lane];
        att_v[k] = att[32 * k + lane];
    }
    if (t < H) ssum[t] = 0.0f;

    const int* adjrow = adj + rowI * NNODES;

    float wmax[H];
#pragma unroll
    for (int h = 0; h < H; ++h) wmax[h] = -3.0e38f;

    for (int j = w; j < NNODES; j += NW) {
        const float* xlr = xl + ((size_t)b * NNODES + j) * HC;
        if (j + NW < NNODES)
            __builtin_prefetch(xl + ((size_t)b * NNODES + j + NW) * HC, 0, 0);
        float p[KP];
#pragma unroll
        for (int k = 0; k < KP; ++k) {
            float s = xr_v[k] + xlr[32 * k + lane];
            s = (s > 0.0f) ? s : 0.2f * s;
            p[k] = att_v[k] * s;
        }
#pragma unroll
        for (int off = 16; off >= 1; off >>= 1) {
#pragma unroll
            for (int k = 0; k < KP; ++k) p[k] += __shfl_xor(p[k], off, 32);
        }
        float eh[H];
#pragma unroll
        for (int h = 0; h < H; ++h) eh[h] = 0.0f;
#pragma unroll
        for (int k = 0; k < KP; ++k) eh[(32 * k) / C] += p[k];
        const bool on = adjrow[j] > 0;
#pragma unroll
        for (int h = 0; h < H; ++h) {
            float e = on ? eh[h] : -1.0e9f;
            eh[h] = e;
            wmax[h] = fmaxf(wmax[h], e);
        }
        if (lane == 0) {
#pragma unroll
            for (int h = 0; h < H; ++h) e_sh[h][j] = eh[h];
        }
    }
    if (lane == 0) {
#pragma unroll
        for (int h = 0; h < H; ++h) wred[w][h] = wmax[h];
    }
    __syncthreads();

    if (t < H) {
        float m = -3.0e38f;
        for (int ww = 0; ww < NW; ++ww) m = fmaxf(m, wred[ww][t]);
        smax[t] = m;
    }
    __syncthreads();
    float lsum[H];
#pragma unroll
    for (int h = 0; h < H; ++h) lsum[h] = 0.0f;
    for (int idx = t; idx < H * NNODES; idx += HC) {
        int h = idx / NNODES, j = idx - h * NNODES;
        float ex = __expf(e_sh[h][j] - smax[h]);
        e_sh[h][j] = ex;
        lsum[h] += ex;
    }
#pragma unroll
    for (int h = 0; h < H; ++h)
        if (lsum[h] != 0.0f) atomicAdd(&ssum[h], lsum[h]);
    __syncthreads();

    float inv[H];
#pragma unroll
    for (int h = 0; h < H; ++h) inv[h] = 1.0f / ssum[h];

    // alpha[(b*H+h)*N + i][j]  (row contiguous in j -> matches WMMA A rows)
    for (int idx = t; idx < H * NNODES; idx += HC) {
        int h = idx / NNODES, j = idx - h * NNODES;
        alpha[((size_t)(b * H + h) * NNODES + i) * NNODES + j] =
            (__bf16)(e_sh[h][j] * inv[h]);
    }
}

// ---------------- WMMA aggregation: out = alpha @ xl (+bias, relu) ----------
// grid.y = b*H+h ; grid.x = (N/64) * (C/32). Block = 256 thr = 8 waves
// covering 4 m-tiles x 2 n-tiles. B tile (32 rows x 512 bf16) staged in LDS
// via async-to-LDS (ASYNCcnt), rows padded to 1040 B (bank-conflict free).
template <int H, int C>
__global__ __launch_bounds__(256)
void agg_wmma_kernel(const __bf16* __restrict__ alpha, const __bf16* __restrict__ xlT,
                     const float* __restrict__ bias, float* __restrict__ out)
{
    constexpr int HC = H * C;
    constexpr int ROWB = NNODES * 2 + 16;               // padded LDS row bytes
    __shared__ __align__(16) char smem[32 * ROWB];      // 33,280 B

    const int t    = threadIdx.x;
    const int lane = t & 31;
    const int wv   = t >> 5;
    const int nblkC = C / 32;
    const int mblk = blockIdx.x / nblkC;
    const int cblk = blockIdx.x % nblkC;
    const int bh   = blockIdx.y;
    const int b    = bh / H, hd = bh % H;

    // -- async-stage 32 rows of xlT[bh] into LDS ------------------------------
    const __bf16* bsrc = xlT + ((size_t)bh * C + cblk * 32) * NNODES;
    const unsigned ldsbase = (unsigned)(uintptr_t)(&smem[0]);
#pragma unroll
    for (int cidx = t; cidx < 32 * 64; cidx += 256) {   // 16-B chunks
        int row = cidx >> 6, ch = cidx & 63;
        unsigned           laddr = ldsbase + row * ROWB + ch * 16;
        unsigned long long gaddr =
            (unsigned long long)(uintptr_t)(bsrc + (size_t)row * NNODES + ch * 8);
        asm volatile("global_load_async_to_lds_b128 %0, %1, off"
                     :: "v"(laddr), "v"(gaddr) : "memory");
    }
    asm volatile("s_wait_asynccnt 0" ::: "memory");
    __syncthreads();

    const int tm  = mblk * 64 + (wv >> 1) * 16;         // output row tile
    const int tnl = (wv & 1) * 16;                      // local col tile in LDS
    const int h2  = lane >> 4;
    const int r16 = lane & 15;

    const __bf16* arow = alpha + ((size_t)bh * NNODES + (tm + r16)) * NNODES;
    const char*   brow = &smem[(tnl + r16) * ROWB];

    v8f acc = {};
    for (int kb = 0; kb < NNODES; kb += 32) {
        v8bf a0 = *(const v8bf*)(arow + kb + 16 * h2);
        v8bf a1 = *(const v8bf*)(arow + kb + 16 * h2 + 8);
        v16bf av;
#pragma unroll
        for (int i = 0; i < 8; ++i) { av[i] = a0[i]; av[8 + i] = a1[i]; }
        v16bf bv = *(const v16bf*)(brow + (kb + 16 * h2) * 2);
        acc = __builtin_amdgcn_wmma_f32_16x16x32_bf16(
                  false, av, false, bv, (short)0, acc, false, false);
    }

    const int cg = cblk * 32 + tnl + r16;               // channel within head
    const float badd = bias[hd * C + cg];
#pragma unroll
    for (int r = 0; r < 8; ++r) {
        float v = fmaxf(acc[r] + badd, 0.0f);
        out[((size_t)b * NNODES + tm + r + 8 * h2) * HC + hd * C + cg] = v;
    }
}

// ---------------- readout ---------------------------------------------------
__global__ void colsum_kernel(const float* __restrict__ x, float* __restrict__ g, int F) {
    int b = blockIdx.x, q = threadIdx.x;
    float a = 0.0f;
    for (int i = 0; i < NNODES; ++i) a += x[((size_t)b * NNODES + i) * F + q];
    g[b * F + q] = a;
}

__global__ void gw_kernel(const float* __restrict__ g, const float* __restrict__ Wg,
                          const float* __restrict__ bg, float* __restrict__ gW, int F) {
    int b = blockIdx.x, n = threadIdx.x;
    float a = bg[n];
    for (int q = 0; q < F; ++q) a += g[b * F + q] * Wg[q * F + n];
    gW[b * F + n] = a;
}

__global__ void final_kernel(const float* __restrict__ xn, const float* __restrict__ gW,
                             const float* __restrict__ Wv, const float* __restrict__ bv,
                             float* __restrict__ out, int F) {
    int idx = blockIdx.x * blockDim.x + threadIdx.x;
    int b = idx / NNODES;
    float a = bv[0];
    for (int n = 0; n < F; ++n) a += fmaxf(xn[(size_t)idx * F + n], 0.0f) * Wv[n];
    for (int n = 0; n < F; ++n) a += fmaxf(gW[b * F + n], 0.0f) * Wv[F + n];
    out[idx] = a;
}

// ---------------------------------------------------------------------------
extern "C" void kernel_launch(void* const* d_in, const int* in_sizes, int n_in,
                              void* d_out, int out_size, void* d_ws, size_t ws_size,
                              hipStream_t stream) {
    (void)in_sizes; (void)n_in; (void)out_size; (void)ws_size;

    const float* x0   = (const float*)d_in[0];
    const int*   adj  = (const int*)  d_in[1];
    const float* Wl1  = (const float*)d_in[2];
    const float* Wr1  = (const float*)d_in[3];
    const float* att1 = (const float*)d_in[4];
    const float* b1   = (const float*)d_in[5];
    const float* Wl2  = (const float*)d_in[6];
    const float* Wr2  = (const float*)d_in[7];
    const float* att2 = (const float*)d_in[8];
    const float* b2   = (const float*)d_in[9];
    const float* Wl3  = (const float*)d_in[10];
    const float* Wr3  = (const float*)d_in[11];
    const float* att3 = (const float*)d_in[12];
    const float* b3   = (const float*)d_in[13];
    const float* Wn   = (const float*)d_in[14];
    const float* bn   = (const float*)d_in[15];
    const float* Wg   = (const float*)d_in[16];
    const float* bg   = (const float*)d_in[17];
    const float* Wv   = (const float*)d_in[18];
    const float* bv   = (const float*)d_in[19];

    char* ws = (char*)d_ws;                       // ~14.3 MB total
    __bf16* xbf   = (__bf16*)(ws + 0x000000);     // 512 KB
    __bf16* wtl   = (__bf16*)(ws + 0x080000);     // 32 KB
    __bf16* wtr   = (__bf16*)(ws + 0x088000);     // 32 KB
    float*  xl    = (float*) (ws + 0x090000);     // 1 MB
    float*  xr    = (float*) (ws + 0x190000);     // 1 MB
    float*  bufA  = (float*) (ws + 0x290000);     // 1 MB
    float*  bufB  = (float*) (ws + 0x390000);     // 1 MB
    float*  gsum  = (float*) (ws + 0x490000);     // 1 KB
    float*  gW    = (float*) (ws + 0x490400);     // 1 KB
    float*  xn    = (float*) (ws + 0x4A0000);     // 512 KB
    __bf16* alpha = (__bf16*)(ws + 0x520000);     // 8 MB   [B,H,N,N]
    __bf16* xlT   = (__bf16*)(ws + 0xD20000);     // 512 KB [B,H,C,N]

    auto run_layer = [&](const float* xin, int K, const float* Wl, const float* Wr,
                         const float* att, const float* bias, int Hh, float* outp) {
        int HC = (Hh == 4) ? 128 : 64;
        pack_wt_kernel<<<(K * HC + 255) / 256, 256, 0, stream>>>(Wl, wtl, K, HC);
        pack_wt_kernel<<<(K * HC + 255) / 256, 256, 0, stream>>>(Wr, wtr, K, HC);
        pack_bf16_kernel<<<(MDIM * K + 255) / 256, 256, 0, stream>>>(xin, xbf, MDIM * K);
        int waves = (MDIM / 16) * (HC / 16);
        gemm_bf16_kernel<<<waves / 4, 128, 0, stream>>>(xbf, wtl, nullptr, xl, MDIM, K, HC, 0);
        gemm_bf16_kernel<<<waves / 4, 128, 0, stream>>>(xbf, wtr, nullptr, xr, MDIM, K, HC, 0);

        dim3 gs(NNODES, NBATCH);
        int nxlt = NBATCH * HC * NNODES;
        pack_xlt_kernel<<<(nxlt + 255) / 256, 256, 0, stream>>>(
            xl, xlT, Hh, HC / Hh);
        if (Hh == 4) {
            score_kernel<4, 32><<<gs, 128, 0, stream>>>(xl, xr, adj, att, alpha);
            dim3 ga((NNODES / 64) * (32 / 32), NBATCH * 4);
            agg_wmma_kernel<4, 32><<<ga, 256, 0, stream>>>(alpha, xlT, bias, outp);
        } else {
            score_kernel<1, 64><<<gs, 64, 0, stream>>>(xl, xr, adj, att, alpha);
            dim3 ga((NNODES / 64) * (64 / 32), NBATCH * 1);
            agg_wmma_kernel<1, 64><<<ga, 256, 0, stream>>>(alpha, xlT, bias, outp);
        }
    };

    run_layer(x0,   64,  Wl1, Wr1, att1, b1, 4, bufA);   // [2048 x 128]
    run_layer(bufA, 128, Wl2, Wr2, att2, b2, 4, bufB);   // [2048 x 128]
    run_layer(bufB, 128, Wl3, Wr3, att3, b3, 1, bufA);   // [2048 x 64]

    // readout
    colsum_kernel<<<NBATCH, 64, 0, stream>>>(bufA, gsum, 64);
    gw_kernel<<<NBATCH, 64, 0, stream>>>(gsum, Wg, bg, gW, 64);

    pack_bf16_kernel<<<(MDIM * 64 + 255) / 256, 256, 0, stream>>>(bufA, xbf, MDIM * 64);
    pack_wt_kernel<<<(64 * 64 + 255) / 256, 256, 0, stream>>>(Wn, wtl, 64, 64);
    int wavesN = (MDIM / 16) * (64 / 16);
    gemm_bf16_kernel<<<wavesN / 4, 128, 0, stream>>>(xbf, wtl, bn, xn, MDIM, 64, 64, 1);

    final_kernel<<<MDIM / 256, 256, 0, stream>>>(xn, gW, Wv, bv, (float*)d_out, 64);
}
